// FusionReasoner_22187801051436
// MI455X (gfx1250) — compile-verified, matrix operand order checked by reading, not run on hardware
//
#include <hip/hip_runtime.h>
#include <hip/hip_bf16.h>

typedef __attribute__((ext_vector_type(16))) _Float16 v16h;
typedef __attribute__((ext_vector_type(8)))  _Float16 v8h;
typedef __attribute__((ext_vector_type(8)))  float    v8f;

// Problem constants
#define BB      4
#define LL      1024
#define MM      (BB * LL)        // 4096 token rows
#define D_EV    128
#define D_RGB   128
#define D_FUSED 384
#define D_REAS  384
#define D_INNER 768
#define D_STATE 16
#define D_CONV  4
#define DT_RANK 24
#define XDBL_N  (DT_RANK + 2 * D_STATE)   // 56
#define LN_EPS  1e-5f

// ---------------------------------------------------------------------------
// f32 -> f16 elementwise convert
__global__ void k_cvt_f16(const float* __restrict__ src, _Float16* __restrict__ dst, int n) {
    int i = blockIdx.x * blockDim.x + threadIdx.x;
    if (i < n) dst[i] = (_Float16)src[i];
}

// W[K][N] f32 row-major  ->  Wt[N][K] f16 row-major (so B fragments load contiguously)
__global__ void k_transpose_f16(const float* __restrict__ W, _Float16* __restrict__ Wt,
                                int K, int N) {
    int i = blockIdx.x * blockDim.x + threadIdx.x;
    if (i >= K * N) return;
    int k = i / N, n = i % N;
    Wt[(size_t)n * K + k] = (_Float16)W[i];
}

// ---------------------------------------------------------------------------
// Fragment loader helper: 16 halves per lane per CDNA5 16-bit A/B 16x32 layout
//   lanes 0-15 : row = r,  halves K = kb+[0..8) then kb+[16..24)   (kb = k0)
//   lanes 16-31: row = r,  halves K = kb+[0..8) then kb+[16..24)   (kb = k0+8)
__device__ __forceinline__ v16h load_frag(const _Float16* rowp, int kb) {
    v8h lo = *(const v8h*)(rowp + kb);
    v8h hi = *(const v8h*)(rowp + kb + 16);
    return __builtin_shufflevector(lo, hi, 0,1,2,3,4,5,6,7,8,9,10,11,12,13,14,15);
}

// Wave-per-macrotile WMMA GEMM: each wave owns a 32x32 output block (2x2 WMMA
// tiles), reusing each A fragment across 2 N-tiles and each B fragment across
// 2 M-tiles -> 4 v_wmma per 8 b128 loads per K-step (2x the arithmetic
// intensity of one-tile-per-wave).
// C[M,N] = A[M,K] (f16) * Bt[N,K]^T (f16) + bias ; M,N % 32 == 0, K % 32 == 0.
__global__ void k_gemm_wmma(const _Float16* __restrict__ A, int lda,
                            const _Float16* __restrict__ Bt, int ldb,
                            const float* __restrict__ bias,
                            float* __restrict__ Cf, int ldcf,
                            _Float16* __restrict__ Ch, int ldch,
                            int M, int N, int K) {
    int wave = threadIdx.x >> 5;
    int lane = threadIdx.x & 31;
    int tile = blockIdx.x * (blockDim.x >> 5) + wave;
    int ntn = N >> 5;
    int ntm = M >> 5;
    if (tile >= ntn * ntm) return;          // wave-uniform: EXEC stays all-ones for WMMA
    int m0 = (tile / ntn) << 5;
    int n0 = (tile % ntn) << 5;
    int r  = lane & 15;
    int hs = lane >> 4;

    const _Float16* a0p = A  + (size_t)(m0 + r)      * lda;
    const _Float16* a1p = A  + (size_t)(m0 + 16 + r) * lda;
    const _Float16* b0p = Bt + (size_t)(n0 + r)      * ldb;
    const _Float16* b1p = Bt + (size_t)(n0 + 16 + r) * ldb;

    v8f acc00 = {}, acc01 = {}, acc10 = {}, acc11 = {};
    for (int k0 = 0; k0 < K; k0 += 32) {
        int kb = k0 + hs * 8;
        v16h av0 = load_frag(a0p, kb);
        v16h av1 = load_frag(a1p, kb);
        v16h bv0 = load_frag(b0p, kb);
        v16h bv1 = load_frag(b1p, kb);
        acc00 = __builtin_amdgcn_wmma_f32_16x16x32_f16(false, av0, false, bv0,
                                                       (short)0, acc00, false, false);
        acc01 = __builtin_amdgcn_wmma_f32_16x16x32_f16(false, av0, false, bv1,
                                                       (short)0, acc01, false, false);
        acc10 = __builtin_amdgcn_wmma_f32_16x16x32_f16(false, av1, false, bv0,
                                                       (short)0, acc10, false, false);
        acc11 = __builtin_amdgcn_wmma_f32_16x16x32_f16(false, av1, false, bv1,
                                                       (short)0, acc11, false, false);
    }

    // D layout: lanes 0-15 -> M = mbase+i, N = lane ; lanes 16-31 -> M = mbase+8+i
    float bb0 = bias ? bias[n0 + r]      : 0.f;
    float bb1 = bias ? bias[n0 + 16 + r] : 0.f;
    int rowb = m0 + hs * 8;
    for (int i = 0; i < 8; ++i) {
        float v00 = acc00[i] + bb0;                 // (rowb+i,    n0+r)
        float v01 = acc01[i] + bb1;                 // (rowb+i,    n0+16+r)
        float v10 = acc10[i] + bb0;                 // (rowb+16+i, n0+r)
        float v11 = acc11[i] + bb1;                 // (rowb+16+i, n0+16+r)
        if (Cf) {
            Cf[(size_t)(rowb + i)      * ldcf + n0 + r]      = v00;
            Cf[(size_t)(rowb + i)      * ldcf + n0 + 16 + r] = v01;
            Cf[(size_t)(rowb + 16 + i) * ldcf + n0 + r]      = v10;
            Cf[(size_t)(rowb + 16 + i) * ldcf + n0 + 16 + r] = v11;
        }
        if (Ch) {
            Ch[(size_t)(rowb + i)      * ldch + n0 + r]      = (_Float16)v00;
            Ch[(size_t)(rowb + i)      * ldch + n0 + 16 + r] = (_Float16)v01;
            Ch[(size_t)(rowb + 16 + i) * ldch + n0 + r]      = (_Float16)v10;
            Ch[(size_t)(rowb + 16 + i) * ldch + n0 + 16 + r] = (_Float16)v11;
        }
    }
}

// ---------------------------------------------------------------------------
// LayerNorm over D_FUSED, one block (128 threads) per row; writes f16 for next GEMM
__global__ void k_layernorm(const float* __restrict__ x, const float* __restrict__ g,
                            const float* __restrict__ bta, _Float16* __restrict__ out) {
    __shared__ float red[128];
    int row = blockIdx.x;
    const float* xr = x + (size_t)row * D_FUSED;

    float s = 0.f;
    for (int i = threadIdx.x; i < D_FUSED; i += 128) s += xr[i];
    red[threadIdx.x] = s; __syncthreads();
    for (int o = 64; o > 0; o >>= 1) {
        if (threadIdx.x < o) red[threadIdx.x] += red[threadIdx.x + o];
        __syncthreads();
    }
    float mu = red[0] / (float)D_FUSED;
    __syncthreads();

    float v = 0.f;
    for (int i = threadIdx.x; i < D_FUSED; i += 128) { float d = xr[i] - mu; v += d * d; }
    red[threadIdx.x] = v; __syncthreads();
    for (int o = 64; o > 0; o >>= 1) {
        if (threadIdx.x < o) red[threadIdx.x] += red[threadIdx.x + o];
        __syncthreads();
    }
    float rs = rsqrtf(red[0] / (float)D_FUSED + LN_EPS);

    for (int i = threadIdx.x; i < D_FUSED; i += 128)
        out[(size_t)row * D_FUSED + i] = (_Float16)((xr[i] - mu) * rs * g[i] + bta[i]);
}

// ---------------------------------------------------------------------------
// Causal depthwise conv (width 4) over time + SiLU; reads u-half of xr, writes f16
__global__ void k_conv_silu(const float* __restrict__ xr, const float* __restrict__ cw,
                            const float* __restrict__ cb, _Float16* __restrict__ u16) {
    int idx = blockIdx.x * blockDim.x + threadIdx.x;     // over MM * D_INNER
    if (idx >= MM * D_INNER) return;
    int c   = idx % D_INNER;
    int row = idx / D_INNER;
    int l   = row % LL;
    float acc = cb[c];
#pragma unroll
    for (int k = 0; k < D_CONV; ++k) {
        int ls = l - (D_CONV - 1) + k;
        if (ls >= 0)
            acc += xr[(size_t)(row - (D_CONV - 1) + k) * (2 * D_INNER) + c] * cw[c * D_CONV + k];
    }
    float s = acc / (1.f + __expf(-acc));                // SiLU
    u16[idx] = (_Float16)s;
}

// ---------------------------------------------------------------------------
// x_dbl = u @ W_x  (K=768, N=56) -> dt[24], Bm[16], Cm[16]; one block per row
__global__ void k_xdbl(const _Float16* __restrict__ u16, const float* __restrict__ Wx,
                       float* __restrict__ dt, float* __restrict__ Bm,
                       float* __restrict__ Cm) {
    __shared__ float us[D_INNER];
    int row = blockIdx.x;
    for (int i = threadIdx.x; i < D_INNER; i += blockDim.x)
        us[i] = (float)u16[(size_t)row * D_INNER + i];
    __syncthreads();
    int j = threadIdx.x;
    if (j < XDBL_N) {
        float acc = 0.f;
        for (int k = 0; k < D_INNER; ++k) acc += us[k] * Wx[k * XDBL_N + j];
        if (j < DT_RANK)                dt[row * DT_RANK + j] = acc;
        else if (j < DT_RANK + D_STATE) Bm[row * D_STATE + (j - DT_RANK)] = acc;
        else                            Cm[row * D_STATE + (j - DT_RANK - D_STATE)] = acc;
    }
}

// ---------------------------------------------------------------------------
// delta = softplus(dt @ W_dt + b_dt); one block per row (256 threads, 3 d each)
__global__ void k_delta(const float* __restrict__ dt, const float* __restrict__ Wdt,
                        const float* __restrict__ bdt, float* __restrict__ delta) {
    __shared__ float ds[DT_RANK];
    int row = blockIdx.x;
    if (threadIdx.x < DT_RANK) ds[threadIdx.x] = dt[row * DT_RANK + threadIdx.x];
    __syncthreads();
    for (int d = threadIdx.x; d < D_INNER; d += blockDim.x) {
        float acc = bdt[d];
#pragma unroll 8
        for (int r = 0; r < DT_RANK; ++r) acc += ds[r] * Wdt[r * D_INNER + d];
        float sp = (acc > 20.f) ? acc : log1pf(__expf(acc));
        delta[(size_t)row * D_INNER + d] = sp;
    }
}

// ---------------------------------------------------------------------------
// Selective scan, sequential over L. 192 blocks x 256 threads.
// thread -> (d = 16 per block, n = 16 states). h kept in a register; y reduced
// over n with wave32 xor-shuffles (16-lane groups); SiLU(res) gate fused.
__global__ void k_scan(const float* __restrict__ delta, const _Float16* __restrict__ u16,
                       const float* __restrict__ Bm, const float* __restrict__ Cm,
                       const float* __restrict__ A_log, const float* __restrict__ Dsk,
                       const float* __restrict__ xr, _Float16* __restrict__ yg) {
    int t  = threadIdx.x;
    int n  = t & 15;
    int dl = t >> 4;                       // 0..15
    int b  = blockIdx.x / (D_INNER / 16);
    int d  = (blockIdx.x % (D_INNER / 16)) * 16 + dl;

    float a   = -__expf(A_log[d * D_STATE + n]);
    float dsk = Dsk[d];
    float h   = 0.f;
    size_t base = (size_t)b * LL;

    for (int l = 0; l < LL; ++l) {
        size_t row = base + l;
        float de = delta[row * D_INNER + d];
        float uv = (float)u16[row * D_INNER + d];
        float bv = Bm[row * D_STATE + n];
        float cv = Cm[row * D_STATE + n];
        h = __expf(de * a) * h + de * bv * uv;
        float p = h * cv;
        p += __shfl_xor(p, 1, 32);
        p += __shfl_xor(p, 2, 32);
        p += __shfl_xor(p, 4, 32);
        p += __shfl_xor(p, 8, 32);
        if (n == 0) {
            float rv = xr[row * (2 * D_INNER) + D_INNER + d];   // res half
            float gv = (p + uv * dsk) * (rv / (1.f + __expf(-rv)));
            yg[row * D_INNER + d] = (_Float16)gv;
        }
    }
}

// ---------------------------------------------------------------------------
extern "C" void kernel_launch(void* const* d_in, const int* in_sizes, int n_in,
                              void* d_out, int out_size, void* d_ws, size_t ws_size,
                              hipStream_t stream) {
    (void)in_sizes; (void)n_in; (void)out_size; (void)ws_size;

    const float* ev     = (const float*)d_in[0];
    const float* rgb    = (const float*)d_in[1];
    const float* W_ev   = (const float*)d_in[2];
    const float* b_ev   = (const float*)d_in[3];
    const float* W_rgb  = (const float*)d_in[4];
    const float* b_rgb  = (const float*)d_in[5];
    const float* ln_g   = (const float*)d_in[6];
    const float* ln_b   = (const float*)d_in[7];
    const float* W_proj = (const float*)d_in[8];
    const float* b_proj = (const float*)d_in[9];
    const float* W_in   = (const float*)d_in[10];
    const float* conv_w = (const float*)d_in[11];
    const float* conv_b = (const float*)d_in[12];
    const float* W_x    = (const float*)d_in[13];
    const float* W_dt   = (const float*)d_in[14];
    const float* b_dt   = (const float*)d_in[15];
    const float* A_log  = (const float*)d_in[16];
    const float* D_skip = (const float*)d_in[17];
    const float* W_out  = (const float*)d_in[18];
    float* out = (float*)d_out;

    char* wp = (char*)d_ws;
    auto alloc = [&](size_t bytes) -> void* {
        void* p = (void*)wp;
        wp += (bytes + 255) & ~(size_t)255;
        return p;
    };
    _Float16* ev16    = (_Float16*)alloc((size_t)MM * D_EV * 2);
    _Float16* rgb16   = (_Float16*)alloc((size_t)MM * D_RGB * 2);
    _Float16* Wt_ev   = (_Float16*)alloc((size_t)(D_FUSED / 2) * D_EV * 2);
    _Float16* Wt_rgb  = (_Float16*)alloc((size_t)(D_FUSED / 2) * D_RGB * 2);
    _Float16* Wt_proj = (_Float16*)alloc((size_t)D_REAS * D_FUSED * 2);
    _Float16* Wt_in   = (_Float16*)alloc((size_t)(2 * D_INNER) * D_REAS * 2);
    _Float16* Wt_out  = (_Float16*)alloc((size_t)D_REAS * D_INNER * 2);
    float*    xf      = (float*)   alloc((size_t)MM * D_FUSED * 4);
    _Float16* xln16   = (_Float16*)alloc((size_t)MM * D_FUSED * 2);
    _Float16* xp16    = (_Float16*)alloc((size_t)MM * D_REAS * 2);
    float*    xrbuf   = (float*)   alloc((size_t)MM * 2 * D_INNER * 4);
    _Float16* u16     = (_Float16*)alloc((size_t)MM * D_INNER * 2);
    float*    dtb     = (float*)   alloc((size_t)MM * DT_RANK * 4);
    float*    Bmb     = (float*)   alloc((size_t)MM * D_STATE * 4);
    float*    Cmb     = (float*)   alloc((size_t)MM * D_STATE * 4);
    float*    deltab  = (float*)   alloc((size_t)MM * D_INNER * 4);
    _Float16* yg16    = (_Float16*)alloc((size_t)MM * D_INNER * 2);

    // ---- prep: converts & weight transposes
    {
        int n = MM * D_EV;
        k_cvt_f16<<<(n + 255) / 256, 256, 0, stream>>>(ev, ev16, n);
        k_cvt_f16<<<(n + 255) / 256, 256, 0, stream>>>(rgb, rgb16, n);
    }
    k_transpose_f16<<<(D_EV * 192 + 255) / 256, 256, 0, stream>>>(W_ev, Wt_ev, D_EV, 192);
    k_transpose_f16<<<(D_RGB * 192 + 255) / 256, 256, 0, stream>>>(W_rgb, Wt_rgb, D_RGB, 192);
    k_transpose_f16<<<(D_FUSED * D_REAS + 255) / 256, 256, 0, stream>>>(W_proj, Wt_proj, D_FUSED, D_REAS);
    k_transpose_f16<<<(D_REAS * 2 * D_INNER + 255) / 256, 256, 0, stream>>>(W_in, Wt_in, D_REAS, 2 * D_INNER);
    k_transpose_f16<<<(D_INNER * D_REAS + 255) / 256, 256, 0, stream>>>(W_out, Wt_out, D_INNER, D_REAS);

    const int TPB = 128;                   // 4 waves per block
    auto gemm_blocks = [](int M, int N) { return ((M >> 5) * (N >> 5) + 3) / 4; };

    // ---- fusion: x[:, :192] = ev @ W_ev + b_ev ; x[:, 192:] = rgb @ W_rgb + b_rgb
    k_gemm_wmma<<<gemm_blocks(MM, 192), TPB, 0, stream>>>(
        ev16, D_EV, Wt_ev, D_EV, b_ev, xf, D_FUSED, (_Float16*)nullptr, 0, MM, 192, D_EV);
    k_gemm_wmma<<<gemm_blocks(MM, 192), TPB, 0, stream>>>(
        rgb16, D_RGB, Wt_rgb, D_RGB, b_rgb, xf + 192, D_FUSED, (_Float16*)nullptr, 0, MM, 192, D_RGB);

    // ---- LayerNorm
    k_layernorm<<<MM, 128, 0, stream>>>(xf, ln_g, ln_b, xln16);

    // ---- proj: xp = LN(x) @ W_proj + b_proj   (f16 out)
    k_gemm_wmma<<<gemm_blocks(MM, D_REAS), TPB, 0, stream>>>(
        xln16, D_FUSED, Wt_proj, D_FUSED, b_proj, (float*)nullptr, 0, xp16, D_REAS,
        MM, D_REAS, D_FUSED);

    // ---- xr = xp @ W_in   (f32 out, [u | res])
    k_gemm_wmma<<<gemm_blocks(MM, 2 * D_INNER), TPB, 0, stream>>>(
        xp16, D_REAS, Wt_in, D_REAS, (const float*)nullptr, xrbuf, 2 * D_INNER,
        (_Float16*)nullptr, 0, MM, 2 * D_INNER, D_REAS);

    // ---- causal depthwise conv + SiLU
    k_conv_silu<<<(MM * D_INNER + 255) / 256, 256, 0, stream>>>(xrbuf, conv_w, conv_b, u16);

    // ---- x_dbl = u @ W_x -> dt, B, C
    k_xdbl<<<MM, 64, 0, stream>>>(u16, W_x, dtb, Bmb, Cmb);

    // ---- delta = softplus(dt @ W_dt + b_dt)
    k_delta<<<MM, 256, 0, stream>>>(dtb, W_dt, b_dt, deltab);

    // ---- selective scan + D-skip + SiLU(res) gate -> yg (f16)
    k_scan<<<BB * (D_INNER / 16), 256, 0, stream>>>(deltab, u16, Bmb, Cmb,
                                                    A_log, D_skip, xrbuf, yg16);

    // ---- out = yg @ W_out  (f32)
    k_gemm_wmma<<<gemm_blocks(MM, D_REAS), TPB, 0, stream>>>(
        yg16, D_INNER, Wt_out, D_INNER, (const float*)nullptr, out, D_REAS,
        (_Float16*)nullptr, 0, MM, D_REAS, D_INNER);
}
